// LlmcFp8Linear_74517682586265
// MI455X (gfx1250) — 8-bit matrix operand order repaired and checked by reading, compiled, not run on hardware
//
#include <hip/hip_runtime.h>
#include <hip/hip_bf16.h>

typedef __attribute__((ext_vector_type(16))) int   v16i;
typedef __attribute__((ext_vector_type(8)))  float v8f;
typedef __attribute__((ext_vector_type(4)))  int   v4i;

typedef __attribute__((address_space(1))) v4i* as1_v4i_p;
typedef __attribute__((address_space(3))) v4i* as3_v4i_p;

#if __has_builtin(__builtin_amdgcn_global_load_async_to_lds_b128) && \
    __has_builtin(__builtin_amdgcn_s_wait_asynccnt)
#define USE_ASYNC_LDS 1
#else
#define USE_ASYNC_LDS 0
#endif

#define FP8_MAX 448.0f

__global__ void act_quant_kernel(const float* __restrict__ x,
                                 unsigned char* __restrict__ q,
                                 float* __restrict__ aT,
                                 int M, int K, int KB) {
    const int lane = threadIdx.x & 31;
    const int warp = threadIdx.x >> 5;
    const long g = (long)blockIdx.x * 8 + warp;
    const int m  = (int)(g / KB);
    const int kb = (int)(g % KB);
    if (m >= M) return;

    const size_t base = (size_t)m * K + (size_t)kb * 128 + lane * 4;
    const float4 xv = *(const float4*)(x + base);

    float amax = fmaxf(fmaxf(fabsf(xv.x), fabsf(xv.y)),
                       fmaxf(fabsf(xv.z), fabsf(xv.w)));
    #pragma unroll
    for (int o = 16; o > 0; o >>= 1)
        amax = fmaxf(amax, __shfl_xor(amax, o, 32));

    const float s = fmaxf(amax, 1e-12f) * (1.0f / FP8_MAX);
    const float r = 1.0f / s;

    int lo = __builtin_amdgcn_cvt_pk_fp8_f32(xv.x * r, xv.y * r, 0, false);
    int dw = __builtin_amdgcn_cvt_pk_fp8_f32(xv.z * r, xv.w * r, lo, true);
    *(unsigned int*)(q + base) = (unsigned int)dw;

    if (lane == 0) aT[(size_t)kb * M + m] = s;
}

__global__ void w_repack_kernel(const float* __restrict__ w,
                                unsigned char* __restrict__ wq,
                                long total) {
    long i = ((long)blockIdx.x * blockDim.x + threadIdx.x) * 8;
    if (i >= total) return;
    const float4 a = *(const float4*)(w + i);
    const float4 b = *(const float4*)(w + i + 4);
    int lo0 = __builtin_amdgcn_cvt_pk_fp8_f32(a.x, a.y, 0, false);
    int d0  = __builtin_amdgcn_cvt_pk_fp8_f32(a.z, a.w, lo0, true);
    int lo1 = __builtin_amdgcn_cvt_pk_fp8_f32(b.x, b.y, 0, false);
    int d1  = __builtin_amdgcn_cvt_pk_fp8_f32(b.z, b.w, lo1, true);
    uint2 o; o.x = (unsigned)d0; o.y = (unsigned)d1;
    *(uint2*)(wq + i) = o;
}

#define BM 128
#define BN 128
#define BK 128
#define BSTRIDE 144

union FragA { v16i v; unsigned u[16]; unsigned long long l[8]; };
union FragB { v16i v; unsigned u[16]; uint4 q4[4]; };

__global__ void __launch_bounds__(256, 1)
fp8_gemm_kernel(const unsigned char* __restrict__ Aq,
                const unsigned char* __restrict__ Wq,
                const float* __restrict__ aT,
                const float* __restrict__ wS,
                const float* __restrict__ bias,
                __hip_bfloat16* __restrict__ out,
                int M, int N, int K, int KB) {
    __shared__ __align__(16) unsigned char ldsB[2][BN * BSTRIDE];

    const int tid  = threadIdx.x;
    const int lane = tid & 31;
    const int warp = tid >> 5;
    const int r16  = lane & 15;
    const int half = lane >> 4;

    const int m0   = blockIdx.y * BM + (warp & 3) * 32;
    const int nw   = (warp >> 2) * 64;
    const int n0   = blockIdx.x * BN + nw;
    const int nblk = blockIdx.x;

    auto stage_B = [&](int kb, int buf) {
        const int kofs = kb * BK;
        #pragma unroll
        for (int i = 0; i < 4; ++i) {
            const int c   = tid + i * 256;
            const int row = c >> 3;
            const int bo  = (c & 7) * 16;
            const unsigned char* g =
                Wq + (size_t)(blockIdx.x * BN + row) * K + kofs + bo;
            unsigned char* l = &ldsB[buf][row * BSTRIDE + bo];
#if USE_ASYNC_LDS
            __builtin_amdgcn_global_load_async_to_lds_b128(
                (as1_v4i_p)g, (as3_v4i_p)l, 0, 0);
#else
            *(uint4*)l = *(const uint4*)g;
#endif
        }
    };

    auto load_fb = [&](FragB& fb, int buf, int nt) {
        const unsigned char* lb = &ldsB[buf][(nw + nt * 16 + r16) * BSTRIDE];
        #pragma unroll
        for (int g = 0; g < 4; ++g)
            fb.q4[g] = *(const uint4*)(lb + 32 * g + 16 * half);
    };

    float acc[2][4][8];
    #pragma unroll
    for (int a = 0; a < 2; ++a)
        #pragma unroll
        for (int b = 0; b < 4; ++b)
            #pragma unroll
            for (int r = 0; r < 8; ++r) acc[a][b][r] = 0.0f;

    const v8f czero = {};

    stage_B(0, 0);

    for (int kb = 0; kb < KB; ++kb) {
        const int buf  = kb & 1;
        const int kofs = kb * BK;

#if USE_ASYNC_LDS
        __builtin_amdgcn_s_wait_asynccnt(0);
#endif
        __syncthreads();

        if (kb + 1 < KB) stage_B(kb + 1, buf ^ 1);

        FragA fa[2];
        #pragma unroll
        for (int mt = 0; mt < 2; ++mt) {
            const unsigned char* ab =
                Aq + (size_t)(m0 + mt * 16 + r16) * K + kofs;
            #pragma unroll
            for (int p = 0; p < 4; ++p) {
                const int kk = (p & 1) * 8 + (p >> 1) * 32 + 16 * half;
                fa[mt].l[p]     = *(const unsigned long long*)(ab + kk);
                fa[mt].l[4 + p] = *(const unsigned long long*)(ab + kk + 64);
            }
        }
        if (kb + 1 < KB)
            __builtin_prefetch(Aq + (size_t)(m0 + r16) * K + kofs + BK, 0, 0);

        const float wSc = wS[(size_t)nblk * KB + kb];
        float cs[2][8];
        #pragma unroll
        for (int mt = 0; mt < 2; ++mt) {
            const float* sp = aT + (size_t)kb * M + m0 + mt * 16 + 8 * half;
            const float4 s0 = *(const float4*)(sp);
            const float4 s1 = *(const float4*)(sp + 4);
            cs[mt][0] = s0.x * wSc; cs[mt][1] = s0.y * wSc;
            cs[mt][2] = s0.z * wSc; cs[mt][3] = s0.w * wSc;
            cs[mt][4] = s1.x * wSc; cs[mt][5] = s1.y * wSc;
            cs[mt][6] = s1.z * wSc; cs[mt][7] = s1.w * wSc;
        }

        v8f d[2][2];
        FragB fbb[2];

        load_fb(fbb[0], buf, 0);
        #pragma unroll
        for (int nt = 0; nt < 4; ++nt) {
            const int p = nt & 1;
            if (nt + 1 < 4) load_fb(fbb[p ^ 1], buf, nt + 1);
            d[p][0] = __builtin_amdgcn_wmma_f32_16x16x128_fp8_fp8(
                fa[0].v, fbb[p].v, (short)0, czero, false, false);
            d[p][1] = __builtin_amdgcn_wmma_f32_16x16x128_fp8_fp8(
                fa[1].v, fbb[p].v, (short)0, czero, false, false);
            if (nt > 0) {
                const int q = p ^ 1;
                #pragma unroll
                for (int mt = 0; mt < 2; ++mt)
                    #pragma unroll
                    for (int r = 0; r < 8; ++r)
                        acc[mt][nt - 1][r] =
                            fmaf(cs[mt][r], d[q][mt][r], acc[mt][nt - 1][r]);
            }
        }
        #pragma unroll
        for (int mt = 0; mt < 2; ++mt)
            #pragma unroll
            for (int r = 0; r < 8; ++r)
                acc[mt][3][r] = fmaf(cs[mt][r], d[1][mt][r], acc[mt][3][r]);
    }

    #pragma unroll
    for (int nt = 0; nt < 4; ++nt) {
        const int col = n0 + nt * 16 + r16;
        const float bv = __bfloat162float(__float2bfloat16(bias[col]));
        #pragma unroll
        for (int mt = 0; mt < 2; ++mt) {
            #pragma unroll
            for (int r = 0; r < 8; ++r) {
                const int row = m0 + mt * 16 + r + 8 * half;
                const float yv =
                    __bfloat162float(__float2bfloat16(acc[mt][nt][r])) + bv;
                out[(size_t)row * N + col] = __float2bfloat16(yv);
            }
        }
    }
}

extern "C" void kernel_launch(void* const* d_in, const int* in_sizes, int n_in,
                              void* d_out, int out_size, void* d_ws, size_t ws_size,
                              hipStream_t stream) {
    const float* x    = (const float*)d_in[0];
    const float* w    = (const float*)d_in[1];
    const float* wsi  = (const float*)d_in[2];
    const float* bias = (const float*)d_in[3];

    const int N  = in_sizes[3];
    const int K  = in_sizes[1] / N;
    const int M  = in_sizes[0] / K;
    const int BS = 128;
    const int KB = K / BS;

    char* ws = (char*)d_ws;
    size_t off = 0;
    unsigned char* Aq = (unsigned char*)(ws + off);
    off += ((size_t)M * K + 255) & ~(size_t)255;
    unsigned char* Wq = (unsigned char*)(ws + off);
    off += ((size_t)N * K + 255) & ~(size_t)255;
    float* aT = (float*)(ws + off);

    {
        const long waves = (long)M * KB;
        dim3 grid((unsigned)((waves + 7) / 8)), block(256);
        act_quant_kernel<<<grid, block, 0, stream>>>(x, Aq, aT, M, K, KB);
    }
    {
        const long total = (long)N * K;
        dim3 grid((unsigned)((total / 8 + 255) / 256)), block(256);
        w_repack_kernel<<<grid, block, 0, stream>>>(w, Wq, total);
    }
    {
        dim3 grid(N / BN, M / BM), block(256);
        fp8_gemm_kernel<<<grid, block, 0, stream>>>(
            Aq, Wq, aT, wsi, bias, (__hip_bfloat16*)d_out, M, N, K, KB);
    }
}
